// EvolveGCN_44822278701843
// MI455X (gfx1250) — compile-verified, matrix operand order checked by reading, not run on hardware
//
#include <hip/hip_runtime.h>
#include <hip/hip_bf16.h>

// ---------------------------------------------------------------------------
// EvolveGCN (2x GCNConv + GRUCell(h0=0) + Linear head) for MI455X / gfx1250
// bf16 WMMA 16x16x32 for all GEMMs (f32 accumulate), weights staged into LDS
// via the Tensor Data Mover (TENSOR_LOAD_TO_LDS + s_wait_tensorcnt), fused
// GRU epilogue, L2-resident edge scatter with hardware f32 atomics.
// ---------------------------------------------------------------------------

#define Nn   100000
#define Ee   1600000
#define INf  165
#define Hh   128
#define KP0  192        // 165 padded up to a multiple of 32

typedef __bf16 v16bf __attribute__((ext_vector_type(16)));
typedef float  v8f   __attribute__((ext_vector_type(8)));
typedef unsigned int u32x4 __attribute__((ext_vector_type(4)));
typedef int          i32x4 __attribute__((ext_vector_type(4)));
typedef int          i32x8 __attribute__((ext_vector_type(8)));

__device__ __forceinline__ void atom_add_f32(float* p, float v) {
    unsafeAtomicAdd(p, v);   // lowers to global_atomic_add_f32 on gfx1250
}

// Issue one TDM 2-D tile load (bf16 elements) Global -> LDS.
// D# layout per CDNA5 ISA ch.8: group0 = {count/lds_addr/global_addr/type},
// group1 = {data_size, tensor dims, tile dims, dim0 stride}; groups 2/3 zero
// (tile_dim2..4 = 0 -> unused) for a 2-D tensor. 6-arg builtin on this
// toolchain: (g0, g1, g2, g3, extra, cpol).
__device__ __forceinline__ void tdm_load_2d_bf16(
    const void* gaddr, unsigned lds_off,
    unsigned rows, unsigned cols, unsigned row_stride /* elements */)
{
    unsigned long long ga = (unsigned long long)gaddr;
    u32x4 g0;
    g0[0] = 1u;                                       // count=1 (valid), user mode
    g0[1] = lds_off;                                  // lds_addr (bytes)
    g0[2] = (unsigned)(ga & 0xFFFFFFFFu);             // global_addr[31:0]
    g0[3] = (unsigned)((ga >> 32) & 0x01FFFFFFu)      // global_addr[56:32]
          | (2u << 30);                               // type = 2 ("image")
    i32x8 g1;
    g1[0] = (int)(1u << 16);                          // data_size=1 -> 2 bytes
    g1[1] = (int)((cols & 0xFFFFu) << 16);            // tensor_dim0[15:0]
    g1[2] = (int)(((cols >> 16) & 0xFFFFu)            // tensor_dim0[31:16]
          | ((rows & 0xFFFFu) << 16));                // tensor_dim1[15:0]
    g1[3] = (int)(((rows >> 16) & 0xFFFFu)            // tensor_dim1[31:16]
          | ((cols & 0xFFFFu) << 16));                // tile_dim0 = cols
    g1[4] = (int)(rows & 0xFFFFu);                    // tile_dim1 = rows, tile_dim2=0
    g1[5] = (int)row_stride;                          // tensor_dim0_stride[31:0]
    g1[6] = 0;                                        // stride[47:32], dim1_stride lo
    g1[7] = 0;
    i32x4 z4 = {};
    i32x8 z8 = {};
    __builtin_amdgcn_tensor_load_to_lds(g0, g1, z4, z4, z8, 0);
}

// ----------------------- elementwise / prep kernels ------------------------

__global__ __launch_bounds__(256) void cast_pad_bf16(
    const float* __restrict__ src, __bf16* __restrict__ dst,
    int rows, int cols, int ldp)
{
    long long i = (long long)blockIdx.x * blockDim.x + threadIdx.x;
    long long total = (long long)rows * ldp;
    if (i >= total) return;
    long long r = i / ldp;
    int c = (int)(i - r * ldp);
    float v = (c < cols) ? src[r * (long long)cols + c] : 0.0f;
    dst[i] = (__bf16)v;
}

__global__ __launch_bounds__(256) void deg_init(float* __restrict__ deg, int n)
{
    int i = blockIdx.x * blockDim.x + threadIdx.x;
    if (i < n) deg[i] = 1.0f;   // self-loop contribution
}

__global__ __launch_bounds__(256) void deg_accum(
    const long long* __restrict__ ei, float* __restrict__ deg, int e)
{
    int i = blockIdx.x * blockDim.x + threadIdx.x;
    if (i >= e) return;
    long long d = ei[(long long)Ee + i];   // dst row of edge_index
    atom_add_f32(&deg[(int)d], 1.0f);
}

__global__ __launch_bounds__(256) void deg_to_dinv(
    const float* __restrict__ deg, float* __restrict__ dinv, int n)
{
    int i = blockIdx.x * blockDim.x + threadIdx.x;
    if (i < n) {
        float d = deg[i];
        dinv[i] = (d > 0.0f) ? rsqrtf(d) : 0.0f;
    }
}

// agg[i][:] = ht[i][:] * dinv[i]^2   (self-loop term; also initializes agg)
__global__ __launch_bounds__(256) void agg_init(
    const float* __restrict__ ht, const float* __restrict__ dinv,
    float* __restrict__ agg, long long total)
{
    long long i = (long long)blockIdx.x * blockDim.x + threadIdx.x;
    if (i >= total) return;
    int row = (int)(i >> 7);
    float d = dinv[row];
    agg[i] = ht[i] * d * d;
}

// one wave per edge: agg[dst][:] += ht[src][:] * dinv[src]*dinv[dst]
__global__ __launch_bounds__(256) void edge_scatter(
    const long long* __restrict__ ei, const float* __restrict__ dinv,
    const float* __restrict__ ht, float* __restrict__ agg, int e)
{
    int gw = blockIdx.x * (blockDim.x >> 5) + (threadIdx.x >> 5);
    if (gw >= e) return;
    int lane = threadIdx.x & 31;
    long long s = ei[gw];
    long long d = ei[(long long)Ee + gw];
    float norm = dinv[(int)s] * dinv[(int)d];
    const float4* hrow = (const float4*)(ht + s * (long long)Hh);
    float4 v = hrow[lane];
    float* arow = agg + d * (long long)Hh + lane * 4;
    atom_add_f32(arow + 0, v.x * norm);
    atom_add_f32(arow + 1, v.y * norm);
    atom_add_f32(arow + 2, v.z * norm);
    atom_add_f32(arow + 3, v.w * norm);
}

// hbf[i][j] = bf16(relu(agg[i][j] + b[j]))
__global__ __launch_bounds__(256) void bias_relu_cast(
    const float* __restrict__ agg, const float* __restrict__ b,
    __bf16* __restrict__ hbf, long long total)
{
    long long i = (long long)blockIdx.x * blockDim.x + threadIdx.x;
    if (i >= total) return;
    int col = (int)(i & (Hh - 1));
    float v = agg[i] + b[col];
    v = v > 0.0f ? v : 0.0f;
    hbf[i] = (__bf16)v;
}

// ----------------------------- WMMA GEMMs ---------------------------------
// A: M x Kp bf16 (row-major, uint4 view, from global). W: 128 x Kp bf16 = Bᵀ,
// DMA'd into LDS once per block by the TDM, fragments read as ds_load_b128.
// Fragment = two 16B quads: dwords kd+0..3 and kd+8..11 (both 16B aligned).

union frag_u { uint4 q[2]; unsigned int u[8]; v16bf v; };

__global__ __launch_bounds__(256) void gcn_gemm_bf16(
    const uint4* __restrict__ A4, const __bf16* __restrict__ Wbf,
    float* __restrict__ C, int M, int Kp)
{
    __shared__ __bf16 wlds[Hh * KP0];                 // 48 KB max
    const int wave = threadIdx.x >> 5;
    if (wave == 0) {
        tdm_load_2d_bf16(Wbf,
                         (unsigned)(uintptr_t)wlds,
                         (unsigned)Hh, (unsigned)Kp, (unsigned)Kp);
        __builtin_amdgcn_s_wait_tensorcnt((short)0);
    }
    __syncthreads();

    const int mi = blockIdx.x * 8 + wave;
    if (mi * 16 >= M) return;                         // after barrier: safe
    const int lane = threadIdx.x & 31;
    const int l15 = lane & 15;
    const int hi  = lane >> 4;
    const int ldA = Kp >> 1;                          // dwords per row
    const size_t arowq = ((size_t)(mi * 16 + l15) * ldA) >> 2;  // uint4 index
    const uint4* wl4 = (const uint4*)wlds;

    v8f acc[8];
#pragma unroll
    for (int t = 0; t < 8; ++t) acc[t] = v8f{};

    for (int k0 = 0; k0 < Kp; k0 += 32) {
        const int kd  = (k0 >> 1) + hi * 4;           // dword offset in row
        const int kq  = kd >> 2;                      // uint4 offset in row
        frag_u af;
        af.q[0] = A4[arowq + kq];
        af.q[1] = A4[arowq + kq + 2];
#pragma unroll
        for (int t = 0; t < 8; ++t) {
            frag_u bf;
            const int wq = (((t * 16 + l15) * ldA) >> 2) + kq;
            bf.q[0] = wl4[wq];
            bf.q[1] = wl4[wq + 2];
            acc[t] = __builtin_amdgcn_wmma_f32_16x16x32_bf16(
                false, af.v, false, bf.v, (short)0, acc[t], false, false);
        }
    }
#pragma unroll
    for (int t = 0; t < 8; ++t) {
#pragma unroll
        for (int r = 0; r < 8; ++r)
            C[(size_t)(mi * 16 + hi * 8 + r) * Hh + t * 16 + l15] = acc[t][r];
    }
}

// Fused GRU(h_prev=0): per wave, 16 rows x one 16-col tile; the three 16x128
// gate weight slabs for this column tile are TDM-loaded into LDS (12 KB).
__global__ __launch_bounds__(256) void gru_gemm_bf16(
    const uint4* __restrict__ A4,           // M x 128 bf16 (uint4 view)
    const __bf16* __restrict__ Wihbf,       // 384 x 128 bf16
    const float* __restrict__ bih, const float* __restrict__ bhh,
    __bf16* __restrict__ hb, float* __restrict__ hf, int M)
{
    __shared__ __bf16 wlds[3 * 16 * Hh];              // 12 KB
    const int wave = threadIdx.x >> 5;
    const int n0  = blockIdx.y * 16;
    if (wave == 0) {
#pragma unroll
        for (int g = 0; g < 3; ++g)
            tdm_load_2d_bf16(Wihbf + (size_t)(g * Hh + n0) * Hh,
                             (unsigned)(uintptr_t)wlds + (unsigned)(g * 16 * Hh * 2),
                             16u, (unsigned)Hh, (unsigned)Hh);
        __builtin_amdgcn_s_wait_tensorcnt((short)0);
    }
    __syncthreads();

    const int mi = blockIdx.x * 8 + wave;
    if (mi * 16 >= M) return;
    const int lane = threadIdx.x & 31;
    const int l15 = lane & 15;
    const int hi  = lane >> 4;
    const size_t arowq = ((size_t)(mi * 16 + l15) * 64) >> 2;   // uint4 index
    const uint4* wl4 = (const uint4*)wlds;

    v8f acc[3];
#pragma unroll
    for (int g = 0; g < 3; ++g) acc[g] = v8f{};

    for (int k0 = 0; k0 < Hh; k0 += 32) {
        const int kd = (k0 >> 1) + hi * 4;
        const int kq = kd >> 2;
        frag_u af;
        af.q[0] = A4[arowq + kq];
        af.q[1] = A4[arowq + kq + 2];
#pragma unroll
        for (int g = 0; g < 3; ++g) {
            frag_u bf;
            const int wq = (((g * 16 + l15) * 64) >> 2) + kq;
            bf.q[0] = wl4[wq];
            bf.q[1] = wl4[wq + 2];
            acc[g] = __builtin_amdgcn_wmma_f32_16x16x32_bf16(
                false, af.v, false, bf.v, (short)0, acc[g], false, false);
        }
    }

    const int n = n0 + l15;
    const float br  = bih[n]          + bhh[n];
    const float bz  = bih[Hh + n]     + bhh[Hh + n];
    const float bin = bih[2 * Hh + n];
    const float bhn = bhh[2 * Hh + n];
#pragma unroll
    for (int r = 0; r < 8; ++r) {
        const size_t row = (size_t)(mi * 16 + hi * 8 + r);
        float rv = 1.0f / (1.0f + __expf(-(acc[0][r] + br)));
        float zv = 1.0f / (1.0f + __expf(-(acc[1][r] + bz)));
        float nv = tanhf(acc[2][r] + bin + rv * bhn);
        float h  = (1.0f - zv) * nv;
        hb[row * Hh + n] = (__bf16)h;
        hf[row * Hh + n] = h;
    }
}

// out[i][0..1] = h[i][:] . Wlin[0..1][:] + blin   (wave per row)
__global__ __launch_bounds__(256) void final_linear(
    const float* __restrict__ h, const float* __restrict__ Wlin,
    const float* __restrict__ blin, float* __restrict__ out, int M)
{
    int gw = blockIdx.x * (blockDim.x >> 5) + (threadIdx.x >> 5);
    if (gw >= M) return;
    int lane = threadIdx.x & 31;
    const float4* hr = (const float4*)(h + (size_t)gw * Hh);
    const float4* w0 = (const float4*)(Wlin);
    const float4* w1 = (const float4*)(Wlin + Hh);
    float4 hv = hr[lane], a = w0[lane], b = w1[lane];
    float p0 = hv.x * a.x + hv.y * a.y + hv.z * a.z + hv.w * a.w;
    float p1 = hv.x * b.x + hv.y * b.y + hv.z * b.z + hv.w * b.w;
#pragma unroll
    for (int off = 16; off; off >>= 1) {
        p0 += __shfl_down(p0, off);
        p1 += __shfl_down(p1, off);
    }
    if (lane == 0) {
        out[(size_t)gw * 2 + 0] = p0 + blin[0];
        out[(size_t)gw * 2 + 1] = p1 + blin[1];
    }
}

// ------------------------------- host side ---------------------------------

static inline int cdiv(long long a, long long b) { return (int)((a + b - 1) / b); }

extern "C" void kernel_launch(void* const* d_in, const int* in_sizes, int n_in,
                              void* d_out, int out_size, void* d_ws, size_t ws_size,
                              hipStream_t stream) {
    const float*     x    = (const float*)d_in[0];
    const long long* ei   = (const long long*)d_in[1];
    const float*     W0   = (const float*)d_in[2];
    const float*     b0   = (const float*)d_in[3];
    const float*     Wih0 = (const float*)d_in[4];
    const float*     bih0 = (const float*)d_in[6];
    const float*     bhh0 = (const float*)d_in[7];
    const float*     W1   = (const float*)d_in[8];
    const float*     b1   = (const float*)d_in[9];
    const float*     Wih1 = (const float*)d_in[10];
    const float*     bih1 = (const float*)d_in[12];
    const float*     bhh1 = (const float*)d_in[13];
    const float*     Wlin = (const float*)d_in[14];
    const float*     blin = (const float*)d_in[15];
    float* out = (float*)d_out;

    // workspace carving (256B aligned)
    char* base = (char*)d_ws;
    size_t off = 0;
    auto carve = [&](size_t bytes) -> char* {
        char* p = base + off;
        off = (off + bytes + 255) & ~(size_t)255;
        return p;
    };
    __bf16* xbf    = (__bf16*)carve((size_t)Nn * KP0 * 2);
    __bf16* w0bf   = (__bf16*)carve((size_t)Hh * KP0 * 2);
    __bf16* wih0bf = (__bf16*)carve((size_t)3 * Hh * Hh * 2);
    __bf16* w1bf   = (__bf16*)carve((size_t)Hh * Hh * 2);
    __bf16* wih1bf = (__bf16*)carve((size_t)3 * Hh * Hh * 2);
    float*  deg    = (float*)carve((size_t)Nn * 4);
    float*  dinv   = (float*)carve((size_t)Nn * 4);
    float*  ht     = (float*)carve((size_t)Nn * Hh * 4);
    float*  agg    = (float*)carve((size_t)Nn * Hh * 4);
    __bf16* hbfA   = (__bf16*)carve((size_t)Nn * Hh * 2);
    __bf16* hbfB   = (__bf16*)carve((size_t)Nn * Hh * 2);
    float*  hf32   = (float*)carve((size_t)Nn * Hh * 4);

    const long long NH = (long long)Nn * Hh;
    const int gemm_blocks = cdiv((long long)Nn / 16, 8);   // 6250 waves / 8 per block

    // --- prep: bf16 casts + degrees ---
    cast_pad_bf16<<<cdiv((long long)Nn * KP0, 256), 256, 0, stream>>>(x, xbf, Nn, INf, KP0);
    cast_pad_bf16<<<cdiv((long long)Hh * KP0, 256), 256, 0, stream>>>(W0, w0bf, Hh, INf, KP0);
    cast_pad_bf16<<<cdiv((long long)3 * Hh * Hh, 256), 256, 0, stream>>>(Wih0, wih0bf, 3 * Hh, Hh, Hh);
    cast_pad_bf16<<<cdiv((long long)Hh * Hh, 256), 256, 0, stream>>>(W1, w1bf, Hh, Hh, Hh);
    cast_pad_bf16<<<cdiv((long long)3 * Hh * Hh, 256), 256, 0, stream>>>(Wih1, wih1bf, 3 * Hh, Hh, Hh);
    deg_init<<<cdiv(Nn, 256), 256, 0, stream>>>(deg, Nn);
    deg_accum<<<cdiv(Ee, 256), 256, 0, stream>>>(ei, deg, Ee);
    deg_to_dinv<<<cdiv(Nn, 256), 256, 0, stream>>>(deg, dinv, Nn);

    // --- layer 0 ---
    gcn_gemm_bf16<<<gemm_blocks, 256, 0, stream>>>((const uint4*)xbf,
                                                   w0bf, ht, Nn, KP0);
    agg_init<<<cdiv(NH, 256), 256, 0, stream>>>(ht, dinv, agg, NH);
    edge_scatter<<<cdiv(Ee, 8), 256, 0, stream>>>(ei, dinv, ht, agg, Ee);
    bias_relu_cast<<<cdiv(NH, 256), 256, 0, stream>>>(agg, b0, hbfA, NH);
    gru_gemm_bf16<<<dim3(gemm_blocks, Hh / 16), 256, 0, stream>>>(
        (const uint4*)hbfA, wih0bf, bih0, bhh0, hbfB, hf32, Nn);

    // --- layer 1 ---
    gcn_gemm_bf16<<<gemm_blocks, 256, 0, stream>>>((const uint4*)hbfB,
                                                   w1bf, ht, Nn, Hh);
    agg_init<<<cdiv(NH, 256), 256, 0, stream>>>(ht, dinv, agg, NH);
    edge_scatter<<<cdiv(Ee, 8), 256, 0, stream>>>(ei, dinv, ht, agg, Ee);
    bias_relu_cast<<<cdiv(NH, 256), 256, 0, stream>>>(agg, b1, hbfA, NH);
    gru_gemm_bf16<<<dim3(gemm_blocks, Hh / 16), 256, 0, stream>>>(
        (const uint4*)hbfA, wih1bf, bih1, bhh1, hbfB, hf32, Nn);

    // --- head ---
    final_linear<<<cdiv(Nn, 8), 256, 0, stream>>>(hf32, Wlin, blin, out, Nn);
}
